// RNNCellWithComplexNeurons_91122026152669
// MI455X (gfx1250) — matrix operand order repaired and checked by reading, not compiled, not run on hardware
//
#include <hip/hip_runtime.h>
#include <math.h>

// ---------------- problem constants ----------------
#define B_ROWS   8192
#define IN_SZ    1024
#define HID      2048
#define NGATES   4096      // ARITY * HIDDEN
#define KTOT     3072      // IN_SZ + HID  (fused K)
#define NUM_TYPES 4
#define CELL_HID 32
#define EPSV     1e-5f

// ---------------- WMMA tile config -----------------
#define BM 256             // workgroup M tile
#define BN 128             // workgroup N tile
#define BK 32              // k-step (bf16)
#define LDT 40             // padded LDS row stride (bf16 elements)
#define NK (KTOT / BK)     // 96 k-steps

typedef __attribute__((ext_vector_type(16))) __bf16 v16bf;
typedef __attribute__((ext_vector_type(8)))  float  v8f;
typedef __attribute__((ext_vector_type(4)))  unsigned int u32x4;

union FragU { u32x4 u[2]; v16bf v; };

#if defined(__has_builtin)
#if __has_builtin(__builtin_amdgcn_tanhf)
#define TANHF(x) __builtin_amdgcn_tanhf(x)
#else
#define TANHF(x) tanhf(x)
#endif
#else
#define TANHF(x) tanhf(x)
#endif

// ---- CDNA5 async global->LDS copy (16B per lane), tracked by ASYNCcnt ----
__device__ __forceinline__ void async_cp16(const __bf16* g, const __bf16* l) {
#if defined(__gfx1250__)
  unsigned lds = (unsigned)(unsigned long long)l;   // low 32 bits = LDS addr
  asm volatile("global_load_async_to_lds_b128 %0, %1, off"
               :: "v"(lds), "v"(g) : "memory");
#else
  *(u32x4*)l = *(const u32x4*)g;
#endif
}

__device__ __forceinline__ void wait_async0() {
#if defined(__gfx1250__)
#if defined(__has_builtin) && __has_builtin(__builtin_amdgcn_s_wait_asynccnt)
  __builtin_amdgcn_s_wait_asynccnt(0);
#else
  asm volatile("s_wait_asynccnt 0x0" ::: "memory");
#endif
#endif
}

// ---------------------------------------------------------------------------
// 1) fp32 -> bf16 pack (writes a column-slice [dstOff, dstOff+srcCols) of dst)
// ---------------------------------------------------------------------------
__global__ __launch_bounds__(256)
void pack_bf16(const float* __restrict__ src, __bf16* __restrict__ dst,
               int rows, int srcCols, int dstLD, int dstOff) {
  long i = (long)blockIdx.x * blockDim.x + threadIdx.x;
  long total = (long)rows * srcCols / 4;
  if (i >= total) return;
  float4 f = ((const float4*)src)[i];
  long e = i * 4;
  int r = (int)(e / srcCols);
  int c = (int)(e - (long)r * srcCols);
  __bf16* d = dst + (long)r * dstLD + dstOff + c;
  d[0] = (__bf16)f.x; d[1] = (__bf16)f.y;
  d[2] = (__bf16)f.z; d[3] = (__bf16)f.w;
}

// ---------------------------------------------------------------------------
// 2) C[m,n] = sum_k A[m,k] * B[n,k]
//    WG tile 256x128, 8 waves in 4x2 grid, wave tile 64x64 = 4x4 WMMA frags.
//    Double-buffered LDS fed by GLOBAL_LOAD_ASYNC_TO_LDS_B128.
// ---------------------------------------------------------------------------
__global__ __launch_bounds__(256)
void gemm_bf16_wmma(const __bf16* __restrict__ A,
                    const __bf16* __restrict__ Bm,
                    float* __restrict__ Cout) {
  __shared__ __bf16 As[2][BM * LDT];   // 2 x 20 KiB
  __shared__ __bf16 Bs[2][BN * LDT];   // 2 x 10 KiB

  const int tid  = threadIdx.x;
  const int lane = tid & 31;
  const int wave = tid >> 5;
  const int wm   = wave >> 1;        // 0..3 -> 64-row slab
  const int wn   = wave & 1;         // 0..1 -> 64-col slab
  const int m    = lane & 15;
  const int h    = lane >> 4;

  const long rowA = (long)blockIdx.y * BM;
  const long rowB = (long)blockIdx.x * BN;

  v8f acc[4][4];
#pragma unroll
  for (int i = 0; i < 4; ++i)
#pragma unroll
    for (int j = 0; j < 4; ++j) acc[i][j] = (v8f)0.0f;

  // async fetch of one BK-slice of both tiles into LDS buffer `buf`
  auto loadTile = [&](int buf, int kt) {
    const long ko = (long)kt * BK;
#pragma unroll
    for (int i = 0; i < 4; ++i) {                 // A: 256 rows * 4 chunks
      int c  = tid + i * 256;                     // 0..1023
      int r  = c >> 2;
      int eo = (c & 3) * 8;
      async_cp16(A + (rowA + r) * (long)KTOT + ko + eo,
                 &As[buf][r * LDT + eo]);
    }
#pragma unroll
    for (int i = 0; i < 2; ++i) {                 // B: 128 rows * 4 chunks
      int c  = tid + i * 256;                     // 0..511
      int r  = c >> 2;
      int eo = (c & 3) * 8;
      async_cp16(Bm + (rowB + r) * (long)KTOT + ko + eo,
                 &Bs[buf][r * LDT + eo]);
    }
  };

  loadTile(0, 0);
  wait_async0();
  __syncthreads();

  for (int kt = 0; kt < NK; ++kt) {
    const int cur = kt & 1;
    if (kt + 1 < NK) loadTile(cur ^ 1, kt + 1);   // overlaps with compute

    // -------- LDS -> fragments (ISA 7.12.2 16-bit layouts) -----------------
    FragU af[4], bfr[4];
#pragma unroll
    for (int i = 0; i < 4; ++i) {
      int r = wm * 64 + i * 16 + m;
      // lanes 0-15 (h=0): dw0-3 = K0..7,  dw4-7 = K16..23
      // lanes16-31 (h=1): dw0-3 = K8..15, dw4-7 = K24..31
      af[i].u[0] = *(const u32x4*)(&As[cur][r * LDT + 8 * h]);
      af[i].u[1] = *(const u32x4*)(&As[cur][r * LDT + 16 + 8 * h]);
    }
#pragma unroll
    for (int j = 0; j < 4; ++j) {
      int r = wn * 64 + j * 16 + m;               // B column n -> W row
      bfr[j].u[0] = *(const u32x4*)(&Bs[cur][r * LDT + 8 * h]);
      bfr[j].u[1] = *(const u32x4*)(&Bs[cur][r * LDT + 16 + 8 * h]);
    }

#pragma unroll
    for (int i = 0; i < 4; ++i)
#pragma unroll
      for (int j = 0; j < 4; ++j)
        acc[i][j] = __builtin_amdgcn_wmma_f32_16x16x32_bf16(
            false, af[i].v, false, bfr[j].v, (short)0, acc[i][j], false, false);

    wait_async0();        // next tile landed in the other buffer
    __syncthreads();      // everyone done reading `cur` before overwrite
  }

  // -------- epilogue: C/D layout — VGPR r: lanes0-15 M=r, lanes16-31 M=8+r --
#pragma unroll
  for (int i = 0; i < 4; ++i)
#pragma unroll
    for (int j = 0; j < 4; ++j)
#pragma unroll
      for (int r = 0; r < 8; ++r) {
        long mm = rowA + wm * 64 + i * 16 + h * 8 + r;
        long nn = rowB + wn * 64 + j * 16 + m;
        Cout[mm * NGATES + nn] = acc[i][j][r];
      }
}

// ---------------------------------------------------------------------------
// 3) Fused bias + LayerNorm + per-type 2->32->1 tanh MLP. One WG per row.
// ---------------------------------------------------------------------------
__global__ __launch_bounds__(256)
void ln_cell(const float* __restrict__ pre,
             const float* __restrict__ bih, const float* __restrict__ bhh,
             const float* __restrict__ w1,  const float* __restrict__ b1,
             const float* __restrict__ w2,  const float* __restrict__ b2,
             float* __restrict__ outAll,    // [B_ROWS, HID]
             float* __restrict__ outLn) {   // [B_ROWS, NGATES]
  __shared__ float red[256];
  __shared__ float w1s[NUM_TYPES * CELL_HID * 2];
  __shared__ float b1s[NUM_TYPES * CELL_HID];
  __shared__ float w2s[NUM_TYPES * CELL_HID];
  __shared__ float b2s[NUM_TYPES];

  const int tid = threadIdx.x;
  if (tid < NUM_TYPES * CELL_HID * 2) w1s[tid] = w1[tid];
  if (tid < NUM_TYPES * CELL_HID) { b1s[tid] = b1[tid]; w2s[tid] = w2[tid]; }
  if (tid < NUM_TYPES) b2s[tid] = b2[tid];

  const long row = blockIdx.x;
  const float* prow = pre + row * (long)NGATES;
  const int base = tid * 16;

  float vals[16];
  float s = 0.f, ss = 0.f;
#pragma unroll
  for (int v = 0; v < 4; ++v) {
    float4 p  = *(const float4*)(prow + base + v * 4);
    float4 bi = *(const float4*)(bih + base + v * 4);
    float4 bh = *(const float4*)(bhh + base + v * 4);
    float a0 = p.x + bi.x + bh.x, a1 = p.y + bi.y + bh.y;
    float a2 = p.z + bi.z + bh.z, a3 = p.w + bi.w + bh.w;
    vals[v*4+0]=a0; vals[v*4+1]=a1; vals[v*4+2]=a2; vals[v*4+3]=a3;
    s += a0 + a1 + a2 + a3;
    ss += a0*a0 + a1*a1 + a2*a2 + a3*a3;
  }

  red[tid] = s; __syncthreads();
  for (int off = 128; off > 0; off >>= 1) {
    if (tid < off) red[tid] += red[tid + off];
    __syncthreads();
  }
  const float mean = red[0] * (1.0f / NGATES);
  __syncthreads();
  red[tid] = ss; __syncthreads();
  for (int off = 128; off > 0; off >>= 1) {
    if (tid < off) red[tid] += red[tid + off];
    __syncthreads();
  }
  const float var  = red[0] * (1.0f / NGATES) - mean * mean;
  const float rstd = rsqrtf(var + EPSV);

  float lnv[16];
#pragma unroll
  for (int u = 0; u < 16; ++u) lnv[u] = (vals[u] - mean) * rstd;

  float* lrow = outLn + row * (long)NGATES + base;
#pragma unroll
  for (int v = 0; v < 4; ++v) {
    float4 o; o.x = lnv[v*4+0]; o.y = lnv[v*4+1]; o.z = lnv[v*4+2]; o.w = lnv[v*4+3];
    *(float4*)(lrow + v * 4) = o;
  }

  // 8 neurons per thread; type block = 1024 gate elements
  const int t = base >> 10;              // 0..3
  const float* w1t = &w1s[t * CELL_HID * 2];
  const float* b1t = &b1s[t * CELL_HID];
  const float* w2t = &w2s[t * CELL_HID];
  const float  bo  = b2s[t];
  float* orow = outAll + row * (long)HID + tid * 8;
#pragma unroll
  for (int u = 0; u < 8; ++u) {
    const float a0 = lnv[2 * u], a1 = lnv[2 * u + 1];
    float o = bo;
#pragma unroll
    for (int c = 0; c < CELL_HID; ++c) {
      float hh = TANHF(fmaf(a1, w1t[c * 2 + 1], fmaf(a0, w1t[c * 2], b1t[c])));
      o = fmaf(hh, w2t[c], o);
    }
    orow[u] = o;
  }
}

// ---------------------------------------------------------------------------
extern "C" void kernel_launch(void* const* d_in, const int* in_sizes, int n_in,
                              void* d_out, int out_size, void* d_ws, size_t ws_size,
                              hipStream_t stream) {
  const float* x   = (const float*)d_in[0];
  const float* hx  = (const float*)d_in[1];
  const float* wih = (const float*)d_in[2];
  const float* whh = (const float*)d_in[3];
  const float* bih = (const float*)d_in[4];
  const float* bhh = (const float*)d_in[5];
  const float* cw1 = (const float*)d_in[6];
  const float* cb1 = (const float*)d_in[7];
  const float* cw2 = (const float*)d_in[8];
  const float* cb2 = (const float*)d_in[9];

  char* ws = (char*)d_ws;
  const size_t szA = (size_t)B_ROWS * KTOT * 2;   // 48 MiB bf16
  const size_t szB = (size_t)NGATES * KTOT * 2;   // 24 MiB bf16
  __bf16* Apack = (__bf16*)(ws);
  __bf16* Bpack = (__bf16*)(ws + szA);
  float*  pre   = (float*)(ws + szA + szB);       // 128 MiB f32

  float* outAll = (float*)d_out;
  float* outLn  = (float*)d_out + (size_t)B_ROWS * HID;

  // pack A = [x | hx], B = [W_ih | W_hh] along K
  {
    int n;
    n = B_ROWS * IN_SZ / 4;
    pack_bf16<<<(n + 255) / 256, 256, 0, stream>>>(x,   Apack, B_ROWS, IN_SZ, KTOT, 0);
    n = B_ROWS * HID / 4;
    pack_bf16<<<(n + 255) / 256, 256, 0, stream>>>(hx,  Apack, B_ROWS, HID,   KTOT, IN_SZ);
    n = NGATES * IN_SZ / 4;
    pack_bf16<<<(n + 255) / 256, 256, 0, stream>>>(wih, Bpack, NGATES, IN_SZ, KTOT, 0);
    n = NGATES * HID / 4;
    pack_bf16<<<(n + 255) / 256, 256, 0, stream>>>(whh, Bpack, NGATES, HID,   KTOT, IN_SZ);
  }

  // pre = A @ B^T  (bias added in epilogue kernel)
  dim3 ggrid(NGATES / BN, B_ROWS / BM);   // 32 x 32
  gemm_bf16_wmma<<<ggrid, 256, 0, stream>>>(Apack, Bpack, pre);

  // fused bias + LN + cell MLP
  ln_cell<<<B_ROWS, 256, 0, stream>>>(pre, bih, bhh, cw1, cb1, cw2, cb2,
                                      outAll, outLn);
}
